// AnchorBasedDetection3DHead_72456098283874
// MI455X (gfx1250) — matrix operand order repaired and checked by reading, not compiled, not run on hardware
//
#include <hip/hip_runtime.h>
#include <hip/hip_bf16.h>
#include <stdint.h>

// ---------------------------------------------------------------------------
// Geometry constants (fixed by the reference problem)
// ---------------------------------------------------------------------------
#define BB 4
#define HH 18
#define WW 80
#define CC 1024            // every conv in this head has C_in = 1024
#define HP (HH + 2)        // padded
#define WP (WW + 2)
#define MTOT (BB * HH * WW)        // 5760 spatial positions
#define FEATPAD_ELEMS ((size_t)BB * HP * WP * CC)      // 6,717,440
#define FEATPAD_BYTES (FEATPAD_ELEMS * 2)

typedef __bf16 bf16;
typedef __attribute__((ext_vector_type(16))) __bf16 v16bf;
typedef __attribute__((ext_vector_type(8)))  __bf16 v8bf;
typedef __attribute__((ext_vector_type(8)))  float  v8f;

__device__ __forceinline__ void mpos(int m, int& b, int& y, int& x) {
  x = m % WW;
  int t = m / WW;
  y = t % HH;
  b = t / HH;
}

// ---------------------------------------------------------------------------
// Utility kernels
// ---------------------------------------------------------------------------
__global__ void zero_kernel(unsigned int* p, size_t n) {
  size_t i = (size_t)blockIdx.x * blockDim.x + threadIdx.x;
  size_t stride = (size_t)gridDim.x * blockDim.x;
  for (; i < n; i += stride) p[i] = 0u;
}

// features NCHW f32 -> padded NHWC bf16 (interior only; halo pre-zeroed)
__global__ void cast_feat_kernel(const float* __restrict__ f, bf16* __restrict__ fp) {
  size_t i = (size_t)blockIdx.x * 256 + threadIdx.x;   // B*H*W*C threads, c fastest
  int c = (int)(i & (CC - 1));
  size_t s = i >> 10;
  int x = (int)(s % WW); s /= WW;
  int y = (int)(s % HH);
  int b = (int)(s / HH);
  float v = f[(((size_t)b * CC + c) * HH + y) * WW + x];
  fp[(((size_t)b * HP + y + 1) * WP + x + 1) * CC + c] = (bf16)v;
}

// w f32 [O][1024][3][3] -> wt bf16 [9][Opad][1024]  (padded region pre-zeroed)
__global__ void prep_w_kernel(const float* __restrict__ w, bf16* __restrict__ wt,
                              int O, int Opad) {
  size_t i = (size_t)blockIdx.x * 256 + threadIdx.x;
  size_t total = (size_t)O * CC * 9;
  if (i >= total) return;
  int c = (int)(i & (CC - 1));
  size_t t = i >> 10;
  int tap = (int)(t % 9);
  int o   = (int)(t / 9);
  wt[(((size_t)tap * Opad + o) << 10) + c] = (bf16)w[((size_t)o * CC + c) * 9 + tap];
}

// per-channel epilogue: y = x*scale + shift   (folds bias (+ eval BN))
__global__ void prep_ss_kernel(const float* __restrict__ bias,
                               const float* __restrict__ g,  const float* __restrict__ be,
                               const float* __restrict__ mn, const float* __restrict__ vr,
                               float* __restrict__ scale, float* __restrict__ shift,
                               int O, int Opad, int bn) {
  int o = blockIdx.x * 256 + threadIdx.x;
  if (o >= Opad) return;
  if (o >= O) { scale[o] = 1.f; shift[o] = 0.f; return; }
  if (bn) {
    float inv = g[o] * rsqrtf(vr[o] + 1e-5f);
    scale[o] = inv;
    shift[o] = bias[o] * inv + be[o] - mn[o] * inv;
  } else {
    scale[o] = 1.f;
    shift[o] = bias[o];
  }
}

// modulated deformable bilinear gather:
//  gval[m][tap][c] = mask(m,tap) * bilinear(featpad, py, px)[c]
__global__ void deform_gather_kernel(const bf16* __restrict__ fp,
                                     const float* __restrict__ offs,   // [m][32]
                                     bf16* __restrict__ gval) {
  int m = blockIdx.x;                 // 0..MTOT-1
  int b, y, x;
  mpos(m, b, y, x);                   // uniform -> SALU
  const float* om = offs + (size_t)m * 32;
  for (int tap = 0; tap < 9; ++tap) {
    int kyy = tap / 3 - 1, kxx = tap % 3 - 1;
    float dy = om[tap], dx = om[9 + tap];
    float mk = 1.f / (1.f + __expf(-om[18 + tap]));
    float py = (float)y + kyy + dy;
    float px = (float)x + kxx + dx;
    float y0 = floorf(py), x0 = floorf(px);
    float wq[4]; size_t aq[4];
    int q = 0;
    for (int dyc = 0; dyc < 2; ++dyc)
      for (int dxc = 0; dxc < 2; ++dxc) {
        float yy = y0 + dyc, xx = x0 + dxc;
        float wgt = (1.f - fabsf(py - yy)) * (1.f - fabsf(px - xx));
        bool valid = (yy >= 0.f) && (yy <= (float)(HH - 1)) &&
                     (xx >= 0.f) && (xx <= (float)(WW - 1));
        int yi = (int)fminf(fmaxf(yy, 0.f), (float)(HH - 1));
        int xi = (int)fminf(fmaxf(xx, 0.f), (float)(WW - 1));
        aq[q] = (((size_t)b * HP + yi + 1) * WP + xi + 1) * CC;
        wq[q] = valid ? wgt * mk : 0.f;
        ++q;
      }
    bf16* gout = gval + ((size_t)m * 9 + tap) * CC;
    for (int c = threadIdx.x; c < CC; c += 256) {
      float v = wq[0] * (float)fp[aq[0] + c] + wq[1] * (float)fp[aq[1] + c] +
                wq[2] * (float)fp[aq[2] + c] + wq[3] * (float)fp[aq[3] + c];
      gout[c] = (bf16)v;
    }
  }
}

// ---------------------------------------------------------------------------
// WMMA implicit-GEMM 3x3 conv (bf16 -> f32 accumulate)
//   AMODE 0: A = padded NHWC features, taps shift (y,x)
//   AMODE 1: A = gval im2col [m][tap][1024]
//   block = 256 thr (8 wave32): 64 spatial rows x 128 out channels
//   each wave: FOUR 16x16 D tiles down M sharing one B fragment, K = 9*1024
//   All div/mod address math done on wave-uniform values (SALU); per-lane
//   terms are simple adds so the hot loop is loads + v_wmma only.
// ---------------------------------------------------------------------------
template<int AMODE, bool RELU, bool OUTF32>
__global__ __launch_bounds__(256, 1)
void conv_wmma_kernel(const bf16* __restrict__ A, const bf16* __restrict__ Wt,
                      const float* __restrict__ scale, const float* __restrict__ shift,
                      void* __restrict__ Out, int Opad, int Ostore) {
  const int lane  = threadIdx.x & 31;
  const int wave  = threadIdx.x >> 5;
  const int ntile = blockIdx.y * 8 + wave;
  if (ntile * 16 >= Opad) return;                 // wave-uniform; EXEC stays all-1s

  const int ml = lane & 15;                       // M (A) or N (B/D) within tile
  const int kh = lane >> 4;                       // K half selector
  const int n  = ntile * 16 + ml;                 // global out channel (B & D column)

  const int m0 = blockIdx.x * 64;                 // 64 spatial rows per block

  // uniform per-tile A bases (SALU div/mod) + one per-lane term
  size_t tbase[4];
#pragma unroll
  for (int t = 0; t < 4; ++t) {
    int mt = m0 + t * 16;                         // uniform
    if (AMODE == 0) {
      int b, y, x;
      mpos(mt, b, y, x);
      tbase[t] = (((size_t)b * HP + y) * WP + x) * CC;
    } else {
      tbase[t] = (size_t)mt * 9 * CC;
    }
  }
  const size_t laneA = (size_t)ml * (AMODE == 0 ? CC : 9 * CC) + (size_t)kh * 16;

  v8f acc[4] = {v8f{}, v8f{}, v8f{}, v8f{}};

  for (int tap = 0; tap < 9; ++tap) {
    size_t aoff;
    if (AMODE == 0) {
      int ky = tap / 3, kx = tap - ky * 3;
      aoff = ((size_t)ky * WP + kx) * CC;
    } else {
      aoff = (size_t)tap * CC;
    }
    const bf16* ap0 = A + tbase[0] + aoff + laneA;
    const bf16* ap1 = A + tbase[1] + aoff + laneA;
    const bf16* ap2 = A + tbase[2] + aoff + laneA;
    const bf16* ap3 = A + tbase[3] + aoff + laneA;
    const bf16* wp  = Wt + (((size_t)tap * Opad + n) << 10) + kh * 16;

    if (tap < 8) {  // prefetch next tap's weight stream (global_prefetch_b8)
      __builtin_prefetch(Wt + (((size_t)(tap + 1) * Opad + n) << 10) + kh * 16, 0, 1);
    }

#pragma unroll 2
    for (int kc = 0; kc < CC / 32; ++kc) {
      union { v16bf v; v8bf h[2]; } a0, a1, a2, a3;
      a0.h[0] = *(const v8bf*)(ap0 + kc * 32);
      a0.h[1] = *(const v8bf*)(ap0 + kc * 32 + 8);
      a1.h[0] = *(const v8bf*)(ap1 + kc * 32);
      a1.h[1] = *(const v8bf*)(ap1 + kc * 32 + 8);
      a2.h[0] = *(const v8bf*)(ap2 + kc * 32);
      a2.h[1] = *(const v8bf*)(ap2 + kc * 32 + 8);
      a3.h[0] = *(const v8bf*)(ap3 + kc * 32);
      a3.h[1] = *(const v8bf*)(ap3 + kc * 32 + 8);
      v16bf bv = *(const v16bf*)(wp + kc * 32);
      acc[0] = __builtin_amdgcn_wmma_f32_16x16x32_bf16(false, a0.v, false, bv,
                                                       (short)0, acc[0], false, false);
      acc[1] = __builtin_amdgcn_wmma_f32_16x16x32_bf16(false, a1.v, false, bv,
                                                       (short)0, acc[1], false, false);
      acc[2] = __builtin_amdgcn_wmma_f32_16x16x32_bf16(false, a2.v, false, bv,
                                                       (short)0, acc[2], false, false);
      acc[3] = __builtin_amdgcn_wmma_f32_16x16x32_bf16(false, a3.v, false, bv,
                                                       (short)0, acc[3], false, false);
    }
  }

  // epilogue: D layout -> lane holds N = lane&15, M = r + 8*(lane>>4).
  // Output base per tile is uniform; per-lane offset computed once, r*stride
  // folds into the store immediate.
  const float sc = scale[n], sh = shift[n];
#pragma unroll
  for (int t = 0; t < 4; ++t) {
    int mt = m0 + t * 16;                         // uniform
    if (OUTF32) {
      float* ob = (float*)Out + (size_t)mt * Ostore + (size_t)kh * 8 * Ostore + n;
#pragma unroll
      for (int r = 0; r < 8; ++r) {
        float v = acc[t][r] * sc + sh;
        if (RELU) v = fmaxf(v, 0.f);
        if (n < Ostore) ob[(size_t)r * Ostore] = v;
      }
    } else {
      int b, y, x;
      mpos(mt, b, y, x);                          // uniform -> SALU
      bf16* ob = (bf16*)Out + (((size_t)b * HP + y + 1) * WP + x + 1) * CC
                 + (size_t)kh * 8 * CC + n;
#pragma unroll
      for (int r = 0; r < 8; ++r) {
        float v = acc[t][r] * sc + sh;
        if (RELU) v = fmaxf(v, 0.f);
        ob[(size_t)r * CC] = (bf16)v;
      }
    }
  }
}

// ---------------------------------------------------------------------------
// Host-side launch
// ---------------------------------------------------------------------------
extern "C" void kernel_launch(void* const* d_in, const int* in_sizes, int n_in,
                              void* d_out, int out_size, void* d_ws, size_t ws_size,
                              hipStream_t stream) {
  (void)in_sizes; (void)n_in; (void)out_size; (void)ws_size;

  const float* feat   = (const float*)d_in[0];
  const float* cls_w1 = (const float*)d_in[1];
  const float* cls_b1 = (const float*)d_in[2];
  const float* cls_w2 = (const float*)d_in[3];
  const float* cls_b2 = (const float*)d_in[4];
  const float* cls_w3 = (const float*)d_in[5];
  const float* cls_b3 = (const float*)d_in[6];
  const float* off_w  = (const float*)d_in[7];
  const float* off_b  = (const float*)d_in[8];
  const float* dcn_w  = (const float*)d_in[9];
  const float* dcn_b  = (const float*)d_in[10];
  const float* bn1_g  = (const float*)d_in[11];
  const float* bn1_b  = (const float*)d_in[12];
  const float* bn1_m  = (const float*)d_in[13];
  const float* bn1_v  = (const float*)d_in[14];
  const float* reg_w2 = (const float*)d_in[15];
  const float* reg_b2 = (const float*)d_in[16];
  const float* bn2_g  = (const float*)d_in[17];
  const float* bn2_b  = (const float*)d_in[18];
  const float* bn2_m  = (const float*)d_in[19];
  const float* bn2_v  = (const float*)d_in[20];
  const float* reg_w3 = (const float*)d_in[21];
  const float* reg_b3 = (const float*)d_in[22];

  // workspace carve-out
  char* ws = (char*)d_ws;
  size_t off = 0;
  auto carve = [&](size_t bytes) {
    void* p = ws + off;
    off += (bytes + 255) & ~(size_t)255;
    return p;
  };
  const size_t WT_BYTES   = (size_t)9 * CC * CC * 2;          // 18.9 MB (max O=1024)
  const size_t OFFS_BYTES = (size_t)MTOT * 32 * 4;
  const size_t GVAL_BYTES = (size_t)MTOT * 9 * CC * 2;

  bf16*  featpad = (bf16*)carve(FEATPAD_BYTES);
  bf16*  actA    = (bf16*)carve(FEATPAD_BYTES);
  bf16*  actB    = (bf16*)carve(FEATPAD_BYTES);
  bf16*  wt      = (bf16*)carve(WT_BYTES);
  float* offs    = (float*)carve(OFFS_BYTES);
  bf16*  gval    = (bf16*)carve(GVAL_BYTES);
  float* scl     = (float*)carve(4096);
  float* shf     = (float*)carve(4096);

  auto zero32 = [&](void* p, size_t bytes) {
    zero_kernel<<<2048, 256, 0, stream>>>((unsigned int*)p, bytes / 4);
  };
  auto prepw = [&](const float* w, int O, int Opad) {
    if (Opad > O) zero32(wt, (size_t)9 * Opad * CC * 2);
    size_t total = (size_t)O * CC * 9;
    prep_w_kernel<<<(unsigned)((total + 255) / 256), 256, 0, stream>>>(w, wt, O, Opad);
  };
  auto prepss = [&](const float* b, const float* g, const float* be,
                    const float* mn, const float* vr, int O, int Opad, int bn) {
    prep_ss_kernel<<<(Opad + 255) / 256, 256, 0, stream>>>(b, g, be, mn, vr,
                                                           scl, shf, O, Opad, bn);
  };

  const int GRID_M = MTOT / 64;  // 90

  // 0) padded bf16 features
  zero32(featpad, FEATPAD_BYTES);
  cast_feat_kernel<<<(unsigned)((size_t)BB * HH * WW * CC / 256), 256, 0, stream>>>(feat, featpad);

  // ---- cls branch ----
  prepw(cls_w1, 1024, 1024);
  prepss(cls_b1, nullptr, nullptr, nullptr, nullptr, 1024, 1024, 0);
  zero32(actA, FEATPAD_BYTES);
  conv_wmma_kernel<0, true, false><<<dim3(GRID_M, 8), 256, 0, stream>>>(
      featpad, wt, scl, shf, actA, 1024, 1024);

  prepw(cls_w2, 1024, 1024);
  prepss(cls_b2, nullptr, nullptr, nullptr, nullptr, 1024, 1024, 0);
  zero32(actB, FEATPAD_BYTES);
  conv_wmma_kernel<0, true, false><<<dim3(GRID_M, 8), 256, 0, stream>>>(
      actA, wt, scl, shf, actB, 1024, 1024);

  prepw(cls_w3, 96, 96);
  prepss(cls_b3, nullptr, nullptr, nullptr, nullptr, 96, 96, 0);
  conv_wmma_kernel<0, false, true><<<dim3(GRID_M, 1), 256, 0, stream>>>(
      actB, wt, scl, shf, (float*)d_out, 96, 96);

  // ---- reg branch: offsets conv (O=27 padded to 32) ----
  prepw(off_w, 27, 32);
  prepss(off_b, nullptr, nullptr, nullptr, nullptr, 27, 32, 0);
  conv_wmma_kernel<0, false, true><<<dim3(GRID_M, 1), 256, 0, stream>>>(
      featpad, wt, scl, shf, offs, 32, 32);

  // bilinear modulated gather -> im2col
  deform_gather_kernel<<<MTOT, 256, 0, stream>>>(featpad, offs, gval);

  // dcn GEMM (+bias+bn1+relu)
  prepw(dcn_w, 1024, 1024);
  prepss(dcn_b, bn1_g, bn1_b, bn1_m, bn1_v, 1024, 1024, 1);
  zero32(actA, FEATPAD_BYTES);
  conv_wmma_kernel<1, true, false><<<dim3(GRID_M, 8), 256, 0, stream>>>(
      gval, wt, scl, shf, actA, 1024, 1024);

  // reg conv2 (+bias+bn2+relu)
  prepw(reg_w2, 1024, 1024);
  prepss(reg_b2, bn2_g, bn2_b, bn2_m, bn2_v, 1024, 1024, 1);
  zero32(actB, FEATPAD_BYTES);
  conv_wmma_kernel<0, true, false><<<dim3(GRID_M, 8), 256, 0, stream>>>(
      actA, wt, scl, shf, actB, 1024, 1024);

  // reg conv3 -> reg_preds
  prepw(reg_w3, 384, 384);
  prepss(reg_b3, nullptr, nullptr, nullptr, nullptr, 384, 384, 0);
  float* reg_out = (float*)d_out + (size_t)BB * HH * WW * 96;
  conv_wmma_kernel<0, false, true><<<dim3(GRID_M, 3), 256, 0, stream>>>(
      actB, wt, scl, shf, reg_out, 384, 384);
}
